// MHADecoder_60404420051397
// MI455X (gfx1250) — matrix operand order repaired and checked by reading, not compiled, not run on hardware
//
#include <hip/hip_runtime.h>
#include <hip/hip_bf16.h>

// ---------------------------------------------------------------------------
// MI455X (gfx1250) MHA-LSTM decoder.
//   - Serial recurrence handled with small per-step kernels on `stream`.
//   - All GEMMs via v_wmma_f32_16x16x32_f16 (wave32 WMMA), f32 accumulate,
//     software-pipelined K-loop (double-buffered fragments) so VMEM latency
//     overlaps the matrix pipe instead of s_wait-stalling every WMMA.
//   - Vocab projection deferred: collect H2[T*B,512] then one 67-GFLOP GEMM.
// ---------------------------------------------------------------------------

typedef __attribute__((ext_vector_type(16))) _Float16 v16h;
typedef __attribute__((ext_vector_type(8)))  float    v8f;

#define Bdim 32
#define Sdim 128
#define Hdim 512
#define Tdim 64
#define Vdim 32000
#define NHEAD 8
#define HD 64

union HFrag {
    v16h v;
    uint4 q[2];   // 2 x 16B = 32B = 16 halves
};

__device__ __forceinline__ void ld_frag(HFrag& f, const _Float16* p0, const _Float16* p1)
{
    f.q[0] = *(const uint4*)p0;
    f.q[1] = *(const uint4*)p1;
}

// ---------------------------------------------------------------------------
// Generic WMMA GEMM:  C[M,N] = A[M,K](f16) * B[N,K](f16)^T + bias[N]
// Each wave computes a 16(M) x 64(N) tile; 8 waves / block.
// K must be a multiple of 64 (true for all call sites: 512, 1536).
// permute!=0 => output row m=(t*B+b) is stored at out[b*T*N + t*N + n].
// ---------------------------------------------------------------------------
__global__ __launch_bounds__(256)
void wmma_gemm_f16(const _Float16* __restrict__ A, const _Float16* __restrict__ B,
                   const float* __restrict__ bias, float* __restrict__ C,
                   int M, int N, int K, int lda, int ldb,
                   int permute, int Tp, int Bp)
{
    const int lane = threadIdx.x & 31;
    const int wave = threadIdx.x >> 5;
    const int wgid = blockIdx.x * 8 + wave;
    const int tilesN = N >> 6;                   // N / 64
    if (wgid >= (M >> 4) * tilesN) return;
    const int tm = wgid / tilesN;
    const int tn = wgid % tilesN;

    const int l16  = lane & 15;
    const int half = lane >> 4;                  // 0 or 1

    // A fragment (16x32 f16): lane l<16 holds K 0..15 of row l,
    // lane l>=16 holds K 16..31 (contiguous, matching B).
    const _Float16* aptr = A + (long)(tm * 16 + l16) * lda + half * 16;
    // B fragment (32x16 f16): lane = N%16, lanes 16-31 hold K 16..31,
    // 16 contiguous K values per lane.
    const _Float16* bbase = B + (long)(tn * 64 + l16) * ldb + half * 16;
    const long bstep = (long)16 * ldb;           // next 16-column B tile

    v8f acc[4] = {};

    // ---- software-pipelined K loop: two fragment buffer sets ----
    HFrag a0, a1;
    HFrag b0[4], b1[4];

    // preload k = 0 into set 0
    ld_frag(a0, aptr, aptr + 8);
#pragma unroll
    for (int j = 0; j < 4; ++j)
        ld_frag(b0[j], bbase + j * bstep, bbase + j * bstep + 8);

    for (int k = 0; k < K; k += 64) {
        const int k1 = k + 32;                   // always < K (K % 64 == 0)
        // issue loads for k+32 into set 1 (overlaps with WMMAs on set 0)
        ld_frag(a1, aptr + k1, aptr + k1 + 8);
#pragma unroll
        for (int j = 0; j < 4; ++j)
            ld_frag(b1[j], bbase + j * bstep + k1, bbase + j * bstep + k1 + 8);
        // prefetch the streamed B operand one stage ahead into near caches
        __builtin_prefetch(bbase + k1 + 96, 0, 2);   // global_prefetch_b8

        // compute on set 0
#pragma unroll
        for (int j = 0; j < 4; ++j)
            acc[j] = __builtin_amdgcn_wmma_f32_16x16x32_f16(
                false, a0.v, false, b0[j].v, (short)0, acc[j], false, false);

        // issue loads for k+64 into set 0 (overlaps with WMMAs on set 1)
        if (k + 64 < K) {
            const int k2 = k + 64;
            ld_frag(a0, aptr + k2, aptr + k2 + 8);
#pragma unroll
            for (int j = 0; j < 4; ++j)
                ld_frag(b0[j], bbase + j * bstep + k2, bbase + j * bstep + k2 + 8);
            __builtin_prefetch(bbase + bstep + k2 + 96, 0, 2);
        }

        // compute on set 1
#pragma unroll
        for (int j = 0; j < 4; ++j)
            acc[j] = __builtin_amdgcn_wmma_f32_16x16x32_f16(
                false, a1.v, false, b1[j].v, (short)0, acc[j], false, false);
    }

    // C/D layout: VGPR e -> row tm*16 + e + 8*half, col = tn*64 + j*16 + l16.
#pragma unroll
    for (int j = 0; j < 4; ++j) {
        const int col = tn * 64 + j * 16 + l16;
        const float bv = bias ? bias[col] : 0.f;
#pragma unroll
        for (int e = 0; e < 8; ++e) {
            const int row = tm * 16 + e + 8 * half;
            long idx;
            if (permute) idx = ((long)(row % Bp) * Tp + (row / Bp)) * (long)N + col;
            else         idx = (long)row * N + col;
            C[idx] = acc[j][e] + bv;
        }
    }
}

// ---------------------------------------------------------------------------
// One-time packing kernels
// ---------------------------------------------------------------------------
__global__ void cvt_f16(const float* __restrict__ s, _Float16* __restrict__ d, int n)
{
    int i = blockIdx.x * 256 + threadIdx.x;
    if (i < n) d[i] = (_Float16)s[i];
}

__global__ void pack_wcat(const float* __restrict__ Wih, const float* __restrict__ Whh,
                          _Float16* __restrict__ Wcat)
{
    int i = blockIdx.x * 256 + threadIdx.x;
    if (i < 2048 * 1536) {
        int n = i / 1536, k = i % 1536;
        float v = (k < 1024) ? Wih[(long)n * 1024 + k] : Whh[(long)n * 512 + (k - 1024)];
        Wcat[i] = (_Float16)v;
    }
}

__global__ void bias_sum(const float* __restrict__ a, const float* __restrict__ b,
                         float* __restrict__ d, int n)
{
    int i = blockIdx.x * 256 + threadIdx.x;
    if (i < n) d[i] = a[i] + b[i];
}

__global__ void init_state(const float* __restrict__ h0, const float* __restrict__ c0,
                           float* __restrict__ h, float* __restrict__ c)
{
    int i = blockIdx.x * 256 + threadIdx.x;
    if (i < Bdim * Hdim) { h[i] = h0[i]; c[i] = c0[i]; }
}

// ---------------------------------------------------------------------------
// Per-step attention + LayerNorm. One block (256 thr = 8 waves) per batch row.
// Writes Acat[b, 0:512]=emb, [512:1024]=ln, [1024:1536]=h_{t-1} as f16.
// ---------------------------------------------------------------------------
__global__ __launch_bounds__(256)
void step_attn(const float* __restrict__ embedding, const int* __restrict__ targets,
               const float* __restrict__ Wq, const float* __restrict__ bq,
               const float* __restrict__ Wo, const float* __restrict__ bo,
               const float* __restrict__ gamma, const float* __restrict__ beta,
               const float* __restrict__ Kh, const float* __restrict__ Vh,
               const float* __restrict__ h_state,
               _Float16* __restrict__ Acat, int t)
{
    __shared__ float s_emb[Hdim], s_q[Hdim], s_att[Hdim], s_x[Hdim];
    __shared__ float s_sc[NHEAD * Sdim];
    __shared__ float s_red[8];

    const int tid = threadIdx.x;
    const int b   = blockIdx.x;
    const int tok = (t == 0) ? 1 : targets[b * Tdim + (t - 1)];   // SOS = 1

    for (int j = tid; j < Hdim; j += 256) {
        float e = embedding[(long)tok * Hdim + j];
        s_emb[j] = e;
        Acat[(long)b * 1536 + j]        = (_Float16)e;
        Acat[(long)b * 1536 + 1024 + j] = (_Float16)h_state[b * Hdim + j];
    }
    __syncthreads();

    // q = emb @ Wq^T + bq
    for (int j = tid; j < Hdim; j += 256) {
        float acc = bq[j];
        const float* w = Wq + (long)j * Hdim;
        for (int k = 0; k < Hdim; ++k) acc += s_emb[k] * w[k];
        s_q[j] = acc;
    }
    __syncthreads();

    // scores[nh][s] = (q_nh . K_{b,s,nh}) * hd^-0.5
    for (int idx = tid; idx < NHEAD * Sdim; idx += 256) {
        int nh = idx >> 7, s = idx & 127;
        const float* kp = Kh + ((long)(b * Sdim + s)) * Hdim + nh * HD;
        const float* qp = s_q + nh * HD;
        float acc = 0.f;
        for (int d = 0; d < HD; ++d) acc += qp[d] * kp[d];
        s_sc[idx] = acc * 0.125f;                // 1/sqrt(64)
    }
    __syncthreads();

    // softmax: wave w owns head w; lane handles 4 of 128 positions (wave32)
    {
        int w = tid >> 5, lane = tid & 31;
        float v0 = s_sc[w * 128 + lane],      v1 = s_sc[w * 128 + 32 + lane];
        float v2 = s_sc[w * 128 + 64 + lane], v3 = s_sc[w * 128 + 96 + lane];
        float m = fmaxf(fmaxf(v0, v1), fmaxf(v2, v3));
        for (int o = 16; o; o >>= 1) m = fmaxf(m, __shfl_xor(m, o, 32));
        v0 = __expf(v0 - m); v1 = __expf(v1 - m);
        v2 = __expf(v2 - m); v3 = __expf(v3 - m);
        float su = v0 + v1 + v2 + v3;
        for (int o = 16; o; o >>= 1) su += __shfl_xor(su, o, 32);
        float inv = 1.f / su;
        s_sc[w * 128 + lane]      = v0 * inv;  s_sc[w * 128 + 32 + lane] = v1 * inv;
        s_sc[w * 128 + 64 + lane] = v2 * inv;  s_sc[w * 128 + 96 + lane] = v3 * inv;
    }
    __syncthreads();

    // att = w . V
    for (int j = tid; j < Hdim; j += 256) {
        int nh = j >> 6, d = j & 63;
        const float* vp = Vh + ((long)(b * Sdim)) * Hdim + nh * HD + d;
        const float* wp = s_sc + nh * Sdim;
        float acc = 0.f;
        for (int s = 0; s < Sdim; ++s) acc += wp[s] * vp[(long)s * Hdim];
        s_att[j] = acc;
    }
    __syncthreads();

    // x = 2 * (att @ Wo^T + bo)   (dropout in eval = identity => x + x)
    for (int j = tid; j < Hdim; j += 256) {
        float acc = bo[j];
        const float* w = Wo + (long)j * Hdim;
        for (int k = 0; k < Hdim; ++k) acc += s_att[k] * w[k];
        s_x[j] = 2.f * acc;
    }
    __syncthreads();

    // LayerNorm over 512
    float part = 0.f;
    for (int j = tid; j < Hdim; j += 256) part += s_x[j];
    for (int o = 16; o; o >>= 1) part += __shfl_xor(part, o, 32);
    if ((tid & 31) == 0) s_red[tid >> 5] = part;
    __syncthreads();
    float mu = (s_red[0] + s_red[1] + s_red[2] + s_red[3] +
                s_red[4] + s_red[5] + s_red[6] + s_red[7]) * (1.f / Hdim);
    __syncthreads();
    part = 0.f;
    for (int j = tid; j < Hdim; j += 256) { float d = s_x[j] - mu; part += d * d; }
    for (int o = 16; o; o >>= 1) part += __shfl_xor(part, o, 32);
    if ((tid & 31) == 0) s_red[tid >> 5] = part;
    __syncthreads();
    float var = (s_red[0] + s_red[1] + s_red[2] + s_red[3] +
                 s_red[4] + s_red[5] + s_red[6] + s_red[7]) * (1.f / Hdim);
    float rstd = rsqrtf(var + 1e-5f);
    for (int j = tid; j < Hdim; j += 256) {
        float ln = (s_x[j] - mu) * rstd * gamma[j] + beta[j];
        Acat[(long)b * 1536 + 512 + j] = (_Float16)ln;
    }
}

// ---------------------------------------------------------------------------
// LSTM pointwise: gates -> (h,c) update, record H2 (f16) for the big GEMM.
// ---------------------------------------------------------------------------
__global__ void lstm_point(const float* __restrict__ gates, float* __restrict__ h_state,
                           float* __restrict__ c_state, _Float16* __restrict__ H2, int t)
{
    int idx = blockIdx.x * 256 + threadIdx.x;
    if (idx >= Bdim * Hdim) return;
    int b = idx >> 9, j = idx & 511;
    const float* g = gates + (long)b * 2048;
    float gi = g[j], gf = g[512 + j], gg = g[1024 + j], go = g[1536 + j];
    float si = 1.f / (1.f + __expf(-gi));
    float sf = 1.f / (1.f + __expf(-gf));
    float so = 1.f / (1.f + __expf(-go));
    float c2 = sf * c_state[idx] + si * tanhf(gg);
    float h2 = so * tanhf(c2);
    c_state[idx] = c2;
    h_state[idx] = h2;
    H2[((long)t * Bdim + b) * Hdim + j] = (_Float16)h2;
}

// ---------------------------------------------------------------------------
extern "C" void kernel_launch(void* const* d_in, const int* in_sizes, int n_in,
                              void* d_out, int out_size, void* d_ws, size_t ws_size,
                              hipStream_t stream)
{
    (void)in_sizes; (void)n_in; (void)out_size; (void)ws_size;

    const float* enc    = (const float*)d_in[0];   // [B,S,H]
    const float* ehid   = (const float*)d_in[1];   // [1,B,H]
    const float* ecell  = (const float*)d_in[2];   // [1,B,H]
    const int*   tgt    = (const int*)  d_in[3];   // [B,T]
    const float* embed  = (const float*)d_in[4];   // [V,H]
    const float* Wq  = (const float*)d_in[5];  const float* bq  = (const float*)d_in[6];
    const float* Wk  = (const float*)d_in[7];  const float* bk  = (const float*)d_in[8];
    const float* Wv  = (const float*)d_in[9];  const float* bv  = (const float*)d_in[10];
    const float* Wo  = (const float*)d_in[11]; const float* bo  = (const float*)d_in[12];
    const float* gma = (const float*)d_in[13]; const float* bta = (const float*)d_in[14];
    const float* Wih = (const float*)d_in[15]; const float* Whh = (const float*)d_in[16];
    const float* bih = (const float*)d_in[17]; const float* bhh = (const float*)d_in[18];
    const float* Wfc = (const float*)d_in[19]; const float* bfc = (const float*)d_in[20];
    float* out = (float*)d_out;                    // [B,T,V] f32

    // ---- workspace carve-out ----
    char* base = (char*)d_ws;
    size_t off = 0;
    auto alloc = [&](size_t bytes) -> void* {
        void* p = base + off;
        off = (off + bytes + 255) & ~(size_t)255;
        return p;
    };
    _Float16* enc_h  = (_Float16*)alloc((size_t)Bdim * Sdim * Hdim * 2);
    _Float16* Wk_h   = (_Float16*)alloc((size_t)Hdim * Hdim * 2);
    _Float16* Wv_h   = (_Float16*)alloc((size_t)Hdim * Hdim * 2);
    _Float16* Wfc_h  = (_Float16*)alloc((size_t)Vdim * Hdim * 2);
    _Float16* Wcat_h = (_Float16*)alloc((size_t)2048 * 1536 * 2);
    float*    Khb    = (float*)   alloc((size_t)Bdim * Sdim * Hdim * 4);
    float*    Vhb    = (float*)   alloc((size_t)Bdim * Sdim * Hdim * 4);
    float*    biasg  = (float*)   alloc((size_t)2048 * 4);
    float*    h_st   = (float*)   alloc((size_t)Bdim * Hdim * 4);
    float*    c_st   = (float*)   alloc((size_t)Bdim * Hdim * 4);
    _Float16* Acat   = (_Float16*)alloc((size_t)Bdim * 1536 * 2);
    float*    gates  = (float*)   alloc((size_t)Bdim * 2048 * 4);
    _Float16* H2     = (_Float16*)alloc((size_t)Tdim * Bdim * Hdim * 2);

    auto cvblk = [](int n) { return (n + 255) / 256; };
    auto gemmBlocks = [](int M, int N) { return ((M / 16) * (N / 64) + 7) / 8; };

    // ---- one-time packing ----
    cvt_f16<<<cvblk(Bdim * Sdim * Hdim), 256, 0, stream>>>(enc, enc_h, Bdim * Sdim * Hdim);
    cvt_f16<<<cvblk(Hdim * Hdim), 256, 0, stream>>>(Wk, Wk_h, Hdim * Hdim);
    cvt_f16<<<cvblk(Hdim * Hdim), 256, 0, stream>>>(Wv, Wv_h, Hdim * Hdim);
    cvt_f16<<<cvblk(Vdim * Hdim), 256, 0, stream>>>(Wfc, Wfc_h, Vdim * Hdim);
    pack_wcat<<<cvblk(2048 * 1536), 256, 0, stream>>>(Wih, Whh, Wcat_h);
    bias_sum<<<cvblk(2048), 256, 0, stream>>>(bih, bhh, biasg, 2048);
    init_state<<<cvblk(Bdim * Hdim), 256, 0, stream>>>(ehid, ecell, h_st, c_st);

    // ---- Kh / Vh : [4096,512] x [512,512]^T  (WMMA) ----
    wmma_gemm_f16<<<gemmBlocks(Bdim * Sdim, Hdim), 256, 0, stream>>>(
        enc_h, Wk_h, bk, Khb, Bdim * Sdim, Hdim, Hdim, Hdim, Hdim, 0, 0, 0);
    wmma_gemm_f16<<<gemmBlocks(Bdim * Sdim, Hdim), 256, 0, stream>>>(
        enc_h, Wv_h, bv, Vhb, Bdim * Sdim, Hdim, Hdim, Hdim, Hdim, 0, 0, 0);

    // ---- serial recurrence ----
    for (int t = 0; t < Tdim; ++t) {
        step_attn<<<Bdim, 256, 0, stream>>>(embed, tgt, Wq, bq, Wo, bo, gma, bta,
                                            Khb, Vhb, h_st, Acat, t);
        // gates[32,2048] = Acat[32,1536] x Wcat[2048,1536]^T + (b_ih+b_hh)
        wmma_gemm_f16<<<gemmBlocks(Bdim, 2048), 256, 0, stream>>>(
            Acat, Wcat_h, biasg, gates, Bdim, 2048, 1536, 1536, 1536, 0, 0, 0);
        lstm_point<<<cvblk(Bdim * Hdim), 256, 0, stream>>>(gates, h_st, c_st, H2, t);
    }

    // ---- vocab projection: [2048,512] x [32000,512]^T, permuted store ----
    wmma_gemm_f16<<<gemmBlocks(Tdim * Bdim, Vdim), 256, 0, stream>>>(
        H2, Wfc_h, bfc, out, Tdim * Bdim, Vdim, Hdim, Hdim, Hdim, 1, Tdim, Bdim);
}
